// DeepEMD_67035849556472
// MI455X (gfx1250) — matrix operand order repaired and checked by reading, not compiled, not run on hardware
//
#include <hip/hip_runtime.h>
#include <hip/hip_bf16.h>

typedef __attribute__((ext_vector_type(16))) _Float16 v16h;
typedef __attribute__((ext_vector_type(8)))  _Float16 v8h;
typedef __attribute__((ext_vector_type(8)))  float    v8f;

#define P_WAY   5
#define KSHOT   1
#define CCH     640
#define HW      49
#define TEMP    12.5f
#define EPSI    0.05f
#define INV_EPS 20.0f
#define ITERS   100

// ---------------- K1: GAP over spatial dim -> gap[b][i][c] ----------------
__global__ void gap_kernel(const float* __restrict__ x, float* __restrict__ gap,
                           int bdim, int nimg, int imgStep, int srcImgs) {
  int idx = blockIdx.x * blockDim.x + threadIdx.x;
  int total = bdim * nimg * CCH;
  if (idx >= total) return;
  int c  = idx % CCH;
  int i  = (idx / CCH) % nimg;
  int bi = idx / (CCH * nimg);
  const float* src = x + (((size_t)bi * srcImgs + (size_t)i * imgStep) * CCH + c) * HW;
  float s = 0.f;
  #pragma unroll
  for (int m = 0; m < HW; ++m) s += src[m];
  gap[idx] = s * (1.0f / HW);
}

// ---- K2: per-(image,spatial) channel-center + L2-normalize, store f16 transposed [b][i][m][C] ----
__global__ void normT_kernel(const float* __restrict__ x, _Float16* __restrict__ outT,
                             int bdim, int nimg, int imgStep, int srcImgs) {
  int wave = threadIdx.x >> 5;
  int lane = threadIdx.x & 31;
  int pos  = blockIdx.x * (blockDim.x >> 5) + wave;
  int total = bdim * nimg * HW;
  if (pos >= total) return;                    // wave-uniform exit
  int m  = pos % HW;
  int i  = (pos / HW) % nimg;
  int bi = pos / (HW * nimg);
  const float* src = x + (((size_t)bi * srcImgs + (size_t)i * imgStep) * CCH) * HW + m;
  float v[CCH / 32];
  float s = 0.f;
  #pragma unroll
  for (int j = 0; j < CCH / 32; ++j) { v[j] = src[(size_t)(lane + 32 * j) * HW]; s += v[j]; }
  #pragma unroll
  for (int off = 16; off > 0; off >>= 1) s += __shfl_xor(s, off, 32);
  float mean = s * (1.0f / CCH);
  float ss = 0.f;
  #pragma unroll
  for (int j = 0; j < CCH / 32; ++j) { v[j] -= mean; ss += v[j] * v[j]; }
  #pragma unroll
  for (int off = 16; off > 0; off >>= 1) ss += __shfl_xor(ss, off, 32);
  float inv = 1.0f / fmaxf(sqrtf(ss), 1e-8f);
  _Float16* dst = outT + (size_t)pos * CCH;    // coalesced: lanes write consecutive halves
  #pragma unroll
  for (int j = 0; j < CCH / 32; ++j) dst[lane + 32 * j] = (_Float16)(v[j] * inv);
}

// ---- K3: fused per-(b,q,p): marginals + WMMA S + Sinkhorn(100) + logit ----
__global__ __launch_bounds__(128) void emd_kernel(
    const float* __restrict__ supRaw, const float* __restrict__ qryRaw,
    const float* __restrict__ supGap, const float* __restrict__ qryGap,
    const _Float16* __restrict__ supT, const _Float16* __restrict__ qryT,
    float* __restrict__ logits, int bdim, int Q) {
  __shared__ float S[HW * HW];
  __shared__ float la[HW], lb[HW], fv[HW], gv[HW];
  __shared__ float red[128];

  int blk = blockIdx.x;                 // bi*Q*P + q*P + p
  int p  = blk % P_WAY;
  int q  = (blk / P_WAY) % Q;
  int bi = blk / (P_WAY * Q);
  int t  = threadIdx.x;

  // ---- phase 1: weight marginals (log_a, log_b) ----
  if (t < HW) {
    int m = t;
    const float* qr = qryRaw + (((size_t)bi * Q + q) * CCH) * HW + m;
    const float* sg = supGap + ((size_t)bi * P_WAY + p) * CCH;
    float acc = 0.f;
    for (int c = 0; c < CCH; ++c) acc += qr[(size_t)c * HW] * sg[c];
    la[m] = fmaxf(acc, 0.f) + (1.0e-3f + 1.0e-5f);
  } else if (t >= 64 && t < 64 + HW) {
    int n = t - 64;
    const float* sr = supRaw + (((size_t)bi * (P_WAY * KSHOT) + p * KSHOT) * CCH) * HW + n;
    const float* qg = qryGap + ((size_t)bi * Q + q) * CCH;
    float acc = 0.f;
    for (int c = 0; c < CCH; ++c) acc += sr[(size_t)c * HW] * qg[c];
    lb[n] = fmaxf(acc, 0.f) + (1.0e-3f + 1.0e-5f);
  }
  __syncthreads();
  if (t == 0)  { float s = 0.f; for (int m = 0; m < HW; ++m) s += la[m]; red[0] = __logf(s); }
  if (t == 64) { float s = 0.f; for (int n = 0; n < HW; ++n) s += lb[n]; red[1] = __logf(s); }
  __syncthreads();
  if (t < HW) { la[t] = __logf(la[t]) - red[0]; lb[t] = __logf(lb[t]) - red[1]; }

  // ---- phase 2: S = qryn^T * supn via v_wmma_f32_16x16x32_f16 (64x64 padded) ----
  {
    int wave = t >> 5, lane = t & 31;
    int row = lane & 15, khalf = lane >> 4;
    const _Float16* qt = qryT + ((size_t)bi * Q + q) * HW * CCH;      // [49][640]
    const _Float16* st = supT + ((size_t)bi * P_WAY + p) * HW * CCH;  // [49][640]
    int mrow = wave * 16 + row;
    int mc = mrow < HW ? mrow : HW - 1;       // clamp pad rows (outputs discarded)
    const _Float16* arow = qt + (size_t)mc * CCH;
    int nclamp[4];
    #pragma unroll
    for (int nt = 0; nt < 4; ++nt) { int n = nt * 16 + row; nclamp[nt] = n < HW ? n : HW - 1; }
    v8f acc[4] = {};
    for (int k0 = 0; k0 < CCH; k0 += 32) {
      // A 16x32 f16 layout: lanes 0-15 hold K {0..7,16..23}, lanes 16-31 hold K {8..15,24..31}
      v8h alo = *(const v8h*)(arow + k0 + 8 * khalf);
      v8h ahi = *(const v8h*)(arow + k0 + 16 + 8 * khalf);
      v16h afrag = __builtin_shufflevector(alo, ahi,
          0,1,2,3,4,5,6,7,8,9,10,11,12,13,14,15);
      #pragma unroll
      for (int nt = 0; nt < 4; ++nt) {
        // B 32x16 f16 layout: lanes 0-15 col N hold K {0..7,16..23}, lanes 16-31 hold K {8..15,24..31}
        const _Float16* bcol = st + (size_t)nclamp[nt] * CCH + k0;
        v8h blo = *(const v8h*)(bcol + 8 * khalf);
        v8h bhi = *(const v8h*)(bcol + 16 + 8 * khalf);
        v16h bfrag = __builtin_shufflevector(blo, bhi,
            0,1,2,3,4,5,6,7,8,9,10,11,12,13,14,15);
        acc[nt] = __builtin_amdgcn_wmma_f32_16x16x32_f16(
            false, afrag, false, bfrag, (short)0, acc[nt], false, false);
      }
    }
    // C/D layout: lanes 0-15: VGPR r -> M=r, N=lane; lanes 16-31: M=r+8, N=lane-16
    #pragma unroll
    for (int nt = 0; nt < 4; ++nt) {
      int ncol2 = nt * 16 + row;
      #pragma unroll
      for (int r = 0; r < 8; ++r) {
        int mrow2 = wave * 16 + khalf * 8 + r;
        if (mrow2 < HW && ncol2 < HW) S[mrow2 * HW + ncol2] = acc[nt][r];
      }
    }
  }
  if (t < HW) { fv[t] = 0.f; gv[t] = 0.f; }
  __syncthreads();

  // ---- phase 3: log-domain Sinkhorn, cost = 1 - S, all in LDS ----
  for (int it = 0; it < ITERS; ++it) {
    if (t < HW) {
      int m = t;
      const float* Sr = S + m * HW;
      float mx = -3.4e38f;
      for (int n = 0; n < HW; ++n) { float x = gv[n] - 1.0f + Sr[n]; mx = fmaxf(mx, x); }
      float s = 0.f;
      for (int n = 0; n < HW; ++n) { float x = gv[n] - 1.0f + Sr[n]; s += __expf((x - mx) * INV_EPS); }
      fv[m] = EPSI * (la[m] - (mx * INV_EPS + __logf(s)));
    }
    __syncthreads();
    if (t < HW) {
      int n = t;
      float mx = -3.4e38f;
      for (int m = 0; m < HW; ++m) { float x = fv[m] - 1.0f + S[m * HW + n]; mx = fmaxf(mx, x); }
      float s = 0.f;
      for (int m = 0; m < HW; ++m) { float x = fv[m] - 1.0f + S[m * HW + n]; s += __expf((x - mx) * INV_EPS); }
      gv[n] = EPSI * (lb[n] - (mx * INV_EPS + __logf(s)));
    }
    __syncthreads();
  }

  // ---- phase 4: logit = TEMP * sum(S * exp((f+g-cost)/eps)) ----
  float accL = 0.f;
  for (int idx = t; idx < HW * HW; idx += 128) {
    int m = idx / HW, n = idx % HW;
    float s = S[idx];
    accL += s * __expf((fv[m] + gv[n] - (1.0f - s)) * INV_EPS);
  }
  red[t] = accL;
  __syncthreads();
  for (int off = 64; off > 0; off >>= 1) { if (t < off) red[t] += red[t + off]; __syncthreads(); }
  if (t == 0) logits[blk] = TEMP * red[0];
}

// ---------------- K4: mean NLL over rows ----------------
__global__ void loss_kernel(const float* __restrict__ logits, const int* __restrict__ labels,
                            float* __restrict__ out, int rows) {
  __shared__ float red[256];
  int t = threadIdx.x;
  float acc = 0.f;
  for (int r = t; r < rows; r += 256) {
    const float* L = logits + (size_t)r * P_WAY;
    float mx = L[0];
    #pragma unroll
    for (int j = 1; j < P_WAY; ++j) mx = fmaxf(mx, L[j]);
    float s = 0.f;
    #pragma unroll
    for (int j = 0; j < P_WAY; ++j) s += __expf(L[j] - mx);
    int lbl = labels[r]; if (lbl < 0) lbl = 0; if (lbl >= P_WAY) lbl = P_WAY - 1;
    acc += (mx + __logf(s)) - L[lbl];
  }
  red[t] = acc;
  __syncthreads();
  for (int off = 128; off > 0; off >>= 1) { if (t < off) red[t] += red[t + off]; __syncthreads(); }
  if (t == 0) out[0] = red[0] / (float)rows;
}

extern "C" void kernel_launch(void* const* d_in, const int* in_sizes, int n_in,
                              void* d_out, int out_size, void* d_ws, size_t ws_size,
                              hipStream_t stream) {
  const float* supRaw = (const float*)d_in[0];
  const float* qryRaw = (const float*)d_in[1];
  const int*   qLab   = (const int*)d_in[3];

  int b = in_sizes[0] / (P_WAY * KSHOT * CCH * HW);   // 8
  int Q = in_sizes[1] / (b * CCH * HW);               // 75

  char* ws = (char*)d_ws;
  size_t off = 0;
  auto wsAlloc = [&](size_t bytes) -> void* {
    off = (off + 255) & ~(size_t)255;
    void* ptr = ws + off;
    off += bytes;
    return ptr;
  };
  float*    supGap = (float*)    wsAlloc((size_t)b * P_WAY * CCH * sizeof(float));
  float*    qryGap = (float*)    wsAlloc((size_t)b * Q     * CCH * sizeof(float));
  _Float16* supT   = (_Float16*) wsAlloc((size_t)b * P_WAY * HW * CCH * sizeof(_Float16));
  _Float16* qryT   = (_Float16*) wsAlloc((size_t)b * Q     * HW * CCH * sizeof(_Float16));
  float*    logits = (float*)    wsAlloc((size_t)b * Q * P_WAY * sizeof(float));

  int tot1 = b * P_WAY * CCH;
  gap_kernel<<<(tot1 + 255) / 256, 256, 0, stream>>>(supRaw, supGap, b, P_WAY, KSHOT, P_WAY * KSHOT);
  int tot2 = b * Q * CCH;
  gap_kernel<<<(tot2 + 255) / 256, 256, 0, stream>>>(qryRaw, qryGap, b, Q, 1, Q);

  int pos1 = b * P_WAY * HW;
  normT_kernel<<<(pos1 + 7) / 8, 256, 0, stream>>>(supRaw, supT, b, P_WAY, KSHOT, P_WAY * KSHOT);
  int pos2 = b * Q * HW;
  normT_kernel<<<(pos2 + 7) / 8, 256, 0, stream>>>(qryRaw, qryT, b, Q, 1, Q);

  emd_kernel<<<b * Q * P_WAY, 128, 0, stream>>>(supRaw, qryRaw, supGap, qryGap,
                                                supT, qryT, logits, b, Q);

  loss_kernel<<<1, 256, 0, stream>>>(logits, qLab, (float*)d_out, b * Q);
}
